// GCNZinc_36515811951203
// MI455X (gfx1250) — matrix operand order repaired and checked by reading, not compiled, not run on hardware
//
#include <hip/hip_runtime.h>
#include <hip/hip_bf16.h>

// ---------------------------------------------------------------------------
// GCN (ZINC-style) on MI455X / gfx1250, wave32.
//   N=50000 nodes, E=800000 edges, G=128 graphs, D=256.
// Strategy:
//   * Layer-1 GEMM folded into a 28-row pre-GEMM (EW1 = emb_table @ W1).
//   * Edge scatter: one wave32 per edge, 8 dims/lane, f32 global atomics (L2).
//   * Layer-2 GEMM: v_wmma_f32_16x16x32_bf16; A tile staged to LDS via the
//     Tensor Data Mover (tensor_load_to_lds + s_wait_tensorcnt), fragments
//     read back with ds_load_b128; B streams from L2.
//   * Pool + MLP: tiny, scalar kernels.
// ---------------------------------------------------------------------------

typedef __attribute__((ext_vector_type(16))) __bf16 v16bf;
typedef __attribute__((ext_vector_type(4)))  __bf16 v4bf;
typedef __attribute__((ext_vector_type(8)))  float  v8f;
typedef unsigned int v4u  __attribute__((ext_vector_type(4)));
typedef int          v8i_t __attribute__((ext_vector_type(8)));
typedef int          v4i_t __attribute__((ext_vector_type(4)));

#define N_NODES_C    50000
#define N_EDGES_C    800000
#define NUM_GRAPHS_C 128
#define DIM_C        256

// ---------------------------------------------------------------- utilities
__global__ void fill_f32(float* __restrict__ p, float v, int n) {
    int i = blockIdx.x * blockDim.x + threadIdx.x;
    if (i < n) p[i] = v;
}

// deg[dst] += 1 per edge (deg pre-filled with 1.0 for self loops)
__global__ void deg_count(const int* __restrict__ dst, float* __restrict__ deg, int e) {
    int i = blockIdx.x * blockDim.x + threadIdx.x;
    if (i < e) atomicAdd(&deg[dst[i]], 1.0f);
}

__global__ void rsqrt_inplace(float* __restrict__ d, int n) {
    int i = blockIdx.x * blockDim.x + threadIdx.x;
    if (i < n) { float v = d[i]; d[i] = v > 0.0f ? rsqrtf(v) : 0.0f; }
}

// EW1[r][n] = sum_k emb[r][k] * W1[k][n]   (28 x 256 @ 256 x 256, tiny)
__global__ void pre_gemm_ew1(const float* __restrict__ emb, const float* __restrict__ W1,
                             float* __restrict__ EW1) {
    int idx = blockIdx.x * blockDim.x + threadIdx.x;   // 28*256 outputs
    if (idx >= 28 * DIM_C) return;
    int r = idx >> 8, n = idx & 255;
    float s = 0.0f;
    const float* er = emb + (size_t)r * DIM_C;
    #pragma unroll 4
    for (int k = 0; k < DIM_C; ++k) s += er[k] * W1[(size_t)k * DIM_C + n];
    EW1[idx] = s;
}

// ---------------------------------------------------------------- scatter
// One wave32 per edge; lane l handles dims l, l+32, ..., l+224.
// use_nx=1: source row is h[node_x[src]] (layer 1, h = EW1, 28 rows, L2-hot).
__global__ __launch_bounds__(256)
void edge_scatter(const float* __restrict__ h, const int* __restrict__ src,
                  const int* __restrict__ dst, const int* __restrict__ node_x,
                  const float* __restrict__ dinv, float* __restrict__ out,
                  int e, int use_nx) {
    int wave = (blockIdx.x * blockDim.x + threadIdx.x) >> 5;
    int lane = threadIdx.x & 31;
    if (wave >= e) return;
    int s = src[wave], d = dst[wave];
    float nrm = dinv[s] * dinv[d];
    int srow = use_nx ? node_x[s] : s;
    const float* hr = h + (size_t)srow * DIM_C;
    float* orow = out + (size_t)d * DIM_C;
    #pragma unroll
    for (int i = 0; i < 8; ++i) {
        int c = lane + i * 32;
        atomicAdd(&orow[c], hr[c] * nrm);
    }
}

// ---------------------------------------------------------------- layer-1 post
// abf = bf16( relu( out1 + EW1[node_x[i]] * dinv[i]^2 + b1 ) ), 4 elems/thread
__global__ void post1_to_bf16(const float* __restrict__ out1, const float* __restrict__ EW1,
                              const int* __restrict__ node_x, const float* __restrict__ dinv,
                              const float* __restrict__ b1, __bf16* __restrict__ abf, int n4) {
    int idx = blockIdx.x * blockDim.x + threadIdx.x;   // n4 = N*D/4
    if (idx >= n4) return;
    int node = idx >> 6;                 // (idx*4) >> 8
    int c4 = (idx & 63) * 4;
    float di = dinv[node];
    float d2 = di * di;
    const float4 o = *(const float4*)(out1 + ((size_t)idx << 2));
    const float4 w = *(const float4*)(EW1 + (size_t)node_x[node] * DIM_C + c4);
    const float4 bb = *(const float4*)(b1 + c4);
    v4bf r;
    r[0] = (__bf16)fmaxf(o.x + w.x * d2 + bb.x, 0.0f);
    r[1] = (__bf16)fmaxf(o.y + w.y * d2 + bb.y, 0.0f);
    r[2] = (__bf16)fmaxf(o.z + w.z * d2 + bb.z, 0.0f);
    r[3] = (__bf16)fmaxf(o.w + w.w * d2 + bb.w, 0.0f);
    *(v4bf*)(abf + ((size_t)idx << 2)) = r;
}

// W2t_bf[n][k] = bf16( W2[k][n] )  — transpose so B-fragment loads are contiguous in K
__global__ void w2_transpose_bf16(const float* __restrict__ W2, __bf16* __restrict__ Wt) {
    int idx = blockIdx.x * blockDim.x + threadIdx.x;
    if (idx >= DIM_C * DIM_C) return;
    int n = idx >> 8, k = idx & 255;
    Wt[idx] = (__bf16)W2[(size_t)k * DIM_C + n];
}

// ---------------------------------------------------------------- WMMA GEMM
// C[M x 256] = A[M x 256] @ W  with W supplied transposed: Wt[n][k] (bf16).
// Block = 8 waves = 2 M-tiles x 4 N-strips (each wave: 16(M) x 64(N) of C).
// The block's 32x256 bf16 A tile (16 KB) is DMA'd to LDS by the Tensor Data
// Mover (one tensor_load_to_lds per block, TENSORcnt-tracked), then fragments
// are read with ds_load_b128. B fragments stream from L2 (128 KB resident).
//
// A frag: lane&15 = row; lanes 0-15 hold K k0..k0+15,
// lanes 16-31 hold K k0+16..k0+31 (contiguous). B frag: lane&15 = col;
// lanes 0-15 hold K k0..k0+15, lanes 16-31 hold K k0+16..k0+31 (contiguous).
__global__ __launch_bounds__(256)
void gemm_bf16_wmma(const __bf16* __restrict__ A, const __bf16* __restrict__ Bt,
                    float* __restrict__ C, int M) {
    __shared__ __bf16 Atile[32 * DIM_C];           // 16 KB
    const int lane = threadIdx.x & 31;
    const int w = threadIdx.x >> 5;                // 0..7
    const int mtLocal = w >> 2;                    // 0 | 1
    const int nt = w & 3;                          // 0..3
    const int half = lane >> 4;                    // 0 | 1
    const int l15 = lane & 15;

    int rowBase = blockIdx.x * 32;
    if (rowBase > M - 32) rowBase = M - 32;        // tail: duplicate tile, benign

    // LDS byte offset of Atile (generic LDS pointer: low 32 bits = LDS address).
    // The ptrtoint escape also tells the compiler the TDM intrinsic may write it.
    unsigned ldsOff = (unsigned)(uintptr_t)(void*)Atile;

    // ---- TDM: async-load A rows [rowBase, rowBase+32) into LDS (wave 0 only)
    if (w == 0) {
        unsigned long long ga =
            (unsigned long long)(const char*)(A + (size_t)rowBase * DIM_C);
        // D# group 0: count=1 | lds_addr | global_addr[56:0] | type=2
        v4u g0 = { 0x1u,
                   ldsOff,
                   (unsigned)(ga & 0xffffffffu),
                   (unsigned)((ga >> 32) & 0x01ffffffu) | (2u << 30) };
        // D# group 1: data_size=2B; tensor_dim0=256; tensor_dim1=32;
        //             tile_dim0=256; tile_dim1=32; tensor_dim0_stride=256
        v8i_t g1 = { (int)(1u << 16),      // [17:16] data_size=1 (2 bytes)
                     (int)(256u << 16),    // tensor_dim0[15:0] << 16
                     (int)(32u << 16),     // tensor_dim0 hi=0 | tensor_dim1 lo
                     (int)(256u << 16),    // tensor_dim1 hi=0 | tile_dim0
                     (int)32,              // tile_dim1=32 | tile_dim2=0
                     (int)256,             // tensor_dim0_stride[31:0]
                     0, 0 };
        v4i_t z4 = { 0, 0, 0, 0 };
#if defined(__clang_major__) && (__clang_major__ >= 23)
        v8i_t z8 = { 0, 0, 0, 0, 0, 0, 0, 0 };
        __builtin_amdgcn_tensor_load_to_lds(g0, g1, z4, z4, z8, 0);
#else
        __builtin_amdgcn_tensor_load_to_lds(g0, g1, z4, z4, 0);
#endif
        __builtin_amdgcn_s_wait_tensorcnt(0);
    }
    __syncthreads();
    // Anchor: LDS contents were produced by the TDM DMA; forbid the compiler
    // from treating Atile as unwritten/undef and deleting the ds_load reads.
    asm volatile("" : : "v"(ldsOff) : "memory");

    const int m0 = rowBase + mtLocal * 16;
    const int n0 = nt * 64;

    v8f acc[4] = {};
    const __bf16* arow = Atile + (size_t)(mtLocal * 16 + l15) * DIM_C;

    #pragma unroll
    for (int k0 = 0; k0 < DIM_C; k0 += 32) {
        union { v16bf v; uint4 q[2]; } a;
        a.q[0] = *(const uint4*)(arow + k0 + half * 16);       // K k0+{0..7 | 16..23}
        a.q[1] = *(const uint4*)(arow + k0 + half * 16 + 8);   // K k0+{8..15 | 24..31}
        #pragma unroll
        for (int j = 0; j < 4; ++j) {
            union { v16bf v; uint4 q[2]; } b;
            const __bf16* brow = Bt + (size_t)(n0 + j * 16 + l15) * DIM_C + k0 + half * 16;
            b.q[0] = *(const uint4*)(brow);
            b.q[1] = *(const uint4*)(brow + 8);
            acc[j] = __builtin_amdgcn_wmma_f32_16x16x32_bf16(
                false, a.v, false, b.v, (short)0, acc[j], false, false);
        }
    }
    // C frag: VGPR r -> row m0 + r + half*8, col n0 + j*16 + l15
    #pragma unroll
    for (int j = 0; j < 4; ++j) {
        #pragma unroll
        for (int r = 0; r < 8; ++r) {
            int row = m0 + r + half * 8;
            int col = n0 + j * 16 + l15;
            C[(size_t)row * DIM_C + col] = acc[j][r];
        }
    }
}

// ---------------------------------------------------------------- layer-2 post + pool
__global__ void graph_count(const int* __restrict__ batch, float* __restrict__ cnt, int n) {
    int i = blockIdx.x * blockDim.x + threadIdx.x;
    if (i < n) atomicAdd(&cnt[batch[i]], 1.0f);
}

// val = relu(out2 + h2*dinv^2 + b2); gsum[batch[node]] += val   (4 elems/thread)
__global__ void post2_pool(const float* __restrict__ out2, const float* __restrict__ h2,
                           const float* __restrict__ dinv, const float* __restrict__ b2,
                           const int* __restrict__ batch, float* __restrict__ gsum, int n4) {
    int idx = blockIdx.x * blockDim.x + threadIdx.x;   // n4 = N*D/4
    if (idx >= n4) return;
    int node = idx >> 6;
    int c4 = (idx & 63) * 4;
    float di = dinv[node];
    float d2 = di * di;
    const float4 o = *(const float4*)(out2 + ((size_t)idx << 2));
    const float4 h = *(const float4*)(h2 + ((size_t)idx << 2));
    const float4 bb = *(const float4*)(b2 + c4);
    float* gr = gsum + (size_t)batch[node] * DIM_C + c4;
    atomicAdd(&gr[0], fmaxf(o.x + h.x * d2 + bb.x, 0.0f));
    atomicAdd(&gr[1], fmaxf(o.y + h.y * d2 + bb.y, 0.0f));
    atomicAdd(&gr[2], fmaxf(o.z + h.z * d2 + bb.z, 0.0f));
    atomicAdd(&gr[3], fmaxf(o.w + h.w * d2 + bb.w, 0.0f));
}

// ---------------------------------------------------------------- MLP head
// g1[g][j] = relu( sum_c relu(gsum[g][c]/cnt) * lin1_W[c][j] + lin1_b[j] )
__global__ void mlp1(const float* __restrict__ gsum, const float* __restrict__ cnt,
                     const float* __restrict__ W, const float* __restrict__ b,
                     float* __restrict__ g1) {
    int g = blockIdx.x;          // 128 graphs
    int j = threadIdx.x;         // 64 hidden
    float ic = 1.0f / fmaxf(cnt[g], 1.0f);
    const float* gr = gsum + (size_t)g * DIM_C;
    float s = b[j];
    #pragma unroll 4
    for (int c = 0; c < DIM_C; ++c)
        s += fmaxf(gr[c] * ic, 0.0f) * W[(size_t)c * 64 + j];
    g1[(size_t)g * 64 + j] = fmaxf(s, 0.0f);
}

__global__ void mlp2(const float* __restrict__ g1, const float* __restrict__ W,
                     const float* __restrict__ b, float* __restrict__ out) {
    int g = threadIdx.x;         // 128
    if (g >= NUM_GRAPHS_C) return;
    const float* gr = g1 + (size_t)g * 64;
    float s = b[0];
    #pragma unroll
    for (int j = 0; j < 64; ++j) s += gr[j] * W[j];
    out[g] = s;
}

// ---------------------------------------------------------------------------
extern "C" void kernel_launch(void* const* d_in, const int* in_sizes, int n_in,
                              void* d_out, int out_size, void* d_ws, size_t ws_size,
                              hipStream_t stream) {
    const int*   node_x  = (const int*)  d_in[0];
    const int*   e_src   = (const int*)  d_in[1];
    const int*   e_dst   = e_src + N_EDGES_C;
    const int*   batch   = (const int*)  d_in[2];
    const float* emb     = (const float*)d_in[3];
    const float* W1      = (const float*)d_in[4];
    const float* b1      = (const float*)d_in[5];
    const float* W2      = (const float*)d_in[6];
    const float* b2      = (const float*)d_in[7];
    const float* lin1_W  = (const float*)d_in[8];
    const float* lin1_b  = (const float*)d_in[9];
    const float* lin2_W  = (const float*)d_in[10];
    const float* lin2_b  = (const float*)d_in[11];
    float* out = (float*)d_out;

    const int N = N_NODES_C, E = N_EDGES_C, G = NUM_GRAPHS_C;
    const size_t ND = (size_t)N * DIM_C;   // 12.8M elements

    // --- workspace carve-up (256B aligned) ---
    char* ws = (char*)d_ws;
    size_t off = 0;
    auto carve = [&](size_t bytes) -> char* {
        off = (off + 255) & ~(size_t)255;
        char* p = ws + off;
        off += bytes;
        return p;
    };
    float*  EW1   = (float*) carve(28 * DIM_C * sizeof(float));      //  28 KB
    float*  dinv  = (float*) carve((size_t)N * sizeof(float));       // 200 KB (deg -> dinv in place)
    float*  outb  = (float*) carve(ND * sizeof(float));              // 51.2 MB (out1, then out2)
    __bf16* abf   = (__bf16*)carve(ND * sizeof(__bf16));             // 25.6 MB
    __bf16* W2t   = (__bf16*)carve((size_t)DIM_C * DIM_C * sizeof(__bf16)); // 128 KB
    float*  h2    = (float*) carve(ND * sizeof(float));              // 51.2 MB
    float*  gsum  = (float*) carve((size_t)G * DIM_C * sizeof(float));
    float*  cnt   = (float*) carve((size_t)G * sizeof(float));
    float*  g1    = (float*) carve((size_t)G * 64 * sizeof(float));

    const int T = 256;
    auto blk = [](size_t n, int t) { return (unsigned)((n + t - 1) / t); };

    // 1) degrees -> dinv
    fill_f32<<<blk(N, T), T, 0, stream>>>(dinv, 1.0f, N);            // self loops
    deg_count<<<blk(E, T), T, 0, stream>>>(e_dst, dinv, E);
    rsqrt_inplace<<<blk(N, T), T, 0, stream>>>(dinv, N);

    // 2) EW1 = emb @ W1  (28 rows)
    pre_gemm_ew1<<<blk(28 * DIM_C, T), T, 0, stream>>>(emb, W1, EW1);

    // 3) layer-1 scatter (gather from 28-row EW1 via node_x), then bias+relu+bf16
    fill_f32<<<blk(ND, T), T, 0, stream>>>(outb, 0.0f, (int)ND);
    edge_scatter<<<blk((size_t)E * 32, T), T, 0, stream>>>(EW1, e_src, e_dst, node_x,
                                                           dinv, outb, E, 1);
    post1_to_bf16<<<blk(ND / 4, T), T, 0, stream>>>(outb, EW1, node_x, dinv, b1,
                                                    abf, (int)(ND / 4));

    // 4) layer-2 GEMM via WMMA bf16 (A tile staged by TDM): h2 = abf @ W2
    w2_transpose_bf16<<<blk(DIM_C * DIM_C, T), T, 0, stream>>>(W2, W2t);
    {
        unsigned blocks = (unsigned)((N / 16 + 1) / 2);   // 1563: 2 M-tiles/block
        gemm_bf16_wmma<<<blocks, T, 0, stream>>>(abf, W2t, h2, N);
    }

    // 5) layer-2 scatter + post + pooling
    fill_f32<<<blk(ND, T), T, 0, stream>>>(outb, 0.0f, (int)ND);
    edge_scatter<<<blk((size_t)E * 32, T), T, 0, stream>>>(h2, e_src, e_dst, node_x,
                                                           dinv, outb, E, 0);
    fill_f32<<<blk((size_t)G * DIM_C, T), T, 0, stream>>>(gsum, 0.0f, G * DIM_C);
    fill_f32<<<blk(G, T), T, 0, stream>>>(cnt, 0.0f, G);
    graph_count<<<blk(N, T), T, 0, stream>>>(batch, cnt, N);
    post2_pool<<<blk(ND / 4, T), T, 0, stream>>>(outb, h2, dinv, b2, batch,
                                                 gsum, (int)(ND / 4));

    // 6) MLP head
    mlp1<<<G, 64, 0, stream>>>(gsum, cnt, lin1_W, lin1_b, g1);
    mlp2<<<1, 128, 0, stream>>>(g1, lin2_W, lin2_b, out);
}
